// AttentionBlock_16630113370602
// MI455X (gfx1250) — matrix operand order repaired and checked by reading, not compiled, not run on hardware
//
#include <hip/hip_runtime.h>
#include <hip/hip_bf16.h>
#include <math.h>

// ---------------------------------------------------------------------------
// AttentionBlock for MI455X (gfx1250, wave32, WMMA).
// - All GEMMs on v_wmma_f32_16x16x32_f16 (fp32 accumulate).
// - Layouts chosen so WMMA fragment loads are 2 x b128 (ISA 7.12.2), except K-major V (scalar LDS reads).
// - B tiles / K,V chunks staged to LDS with global_load_async_to_lds_b128
//   using ping-pong double buffering: async issue for chunk i+1 overlaps the
//   WMMA compute on chunk i; one workgroup barrier per iteration.
// - Flash-style attention: 4096x4096 scores never touch HBM.
// ---------------------------------------------------------------------------

typedef _Float16 f16;
typedef __attribute__((ext_vector_type(16))) _Float16 v16h;
typedef __attribute__((ext_vector_type(8)))  _Float16 v8h;
typedef __attribute__((ext_vector_type(8)))  float    v8f;

#define B_    2
#define C_    512
#define HW_   4096
#define NH_   8
#define HD_   64
#define G_    32
#define CPG_  16

#define WMMA_F16(a, b, c) \
  __builtin_amdgcn_wmma_f32_16x16x32_f16(false, (a), false, (b), (short)0, (c), false, false)

// ---- CDNA5 async copy: 16B global -> LDS, tracked by ASYNCcnt --------------
__device__ __forceinline__ void async_ld_b128(const void* gptr, void* lptr) {
  unsigned loff = (unsigned)(size_t)lptr;   // generic LDS ptr low 32b = LDS offset
  asm volatile("global_load_async_to_lds_b128 %0, %1, off"
               :: "v"(loff), "v"(gptr) : "memory");
}
__device__ __forceinline__ void wait_async0() {
#if __has_builtin(__builtin_amdgcn_s_wait_asynccnt)
  __builtin_amdgcn_s_wait_asynccnt(0);
#else
  asm volatile("s_wait_asynccnt 0x0" ::: "memory");
#endif
}

// ---- fragment loaders (2 x 16B, except K-major B: 16 scalar reads) ---------------------------------------
// A fragment 16x32 (MxK) from row-major source, leading dim ld (elements).
__device__ __forceinline__ v16h ld_a16x32(const f16* base, int ld) {
  const int lane = threadIdx.x & 31;
  const int row = lane & 15, half = lane >> 4;
  const f16* p = base + row * ld + (half << 4);
  v8h lo = *(const v8h*)(p);        // k = 16h .. 16h+7
  v8h hi = *(const v8h*)(p + 8);    // k = 16h+8 .. 16h+15
  return __builtin_shufflevector(lo, hi, 0,1,2,3,4,5,6,7,8,9,10,11,12,13,14,15);
}

// B fragment 32x16 (KxN) from N-major source: element (k,col) at base[col*ld+k].
__device__ __forceinline__ v16h ld_b32x16_nmajor(const f16* base, int ld) {
  const int lane = threadIdx.x & 31;
  const int col = lane & 15, half = lane >> 4;
  const f16* p = base + col * ld + (half << 4);
  v8h lo = *(const v8h*)(p);        // k = 16h .. 16h+7
  v8h hi = *(const v8h*)(p + 8);    // k = 16h+8 .. 16h+15
  return __builtin_shufflevector(lo, hi, 0,1,2,3,4,5,6,7,8,9,10,11,12,13,14,15);
}

// B fragment 32x16 (KxN) from K-major source (LDS): element (k,col) at base[k*ld+col].
__device__ __forceinline__ v16h ld_b32x16_kmajor(const f16* base, int ld) {
  const int lane = threadIdx.x & 31;
  const int col = lane & 15, half = lane >> 4;
  v16h b;
#pragma unroll
  for (int v = 0; v < 8; ++v) {
    const int k = (half << 4) + (v << 1);
    b[2 * v]     = base[(k)     * ld + col];
    b[2 * v + 1] = base[(k + 1) * ld + col];
  }
  return b;
}

// ---------------------------------------------------------------------------
// Kernel 0: fp32 -> f16 weight conversion
// ---------------------------------------------------------------------------
__global__ void k_convert_weights(const float* __restrict__ qkv_w,
                                  const float* __restrict__ proj_w,
                                  f16* __restrict__ wq, f16* __restrict__ wp) {
  const int stride = gridDim.x * blockDim.x;
  for (int t = blockIdx.x * blockDim.x + threadIdx.x; t < 3 * C_ * C_; t += stride)
    wq[t] = (f16)qkv_w[t];
  for (int t = blockIdx.x * blockDim.x + threadIdx.x; t < C_ * C_; t += stride)
    wp[t] = (f16)proj_w[t];
}

// ---------------------------------------------------------------------------
// Kernel 1: GroupNorm, writing h transposed to [b, n, c] (N-major for GEMM B).
// ---------------------------------------------------------------------------
__global__ void __launch_bounds__(256) k_groupnorm(const float* __restrict__ x,
                                                   const float* __restrict__ gw,
                                                   const float* __restrict__ gb,
                                                   f16* __restrict__ ht) {
  __shared__ float red[2][256];
  __shared__ __align__(16) _Float16 tile[CPG_][264];  // 16c x 256n (+8 pad)
  const int bg = blockIdx.x;                // 0..63
  const int b = bg / G_, g = bg % G_;
  const float* xb = x + (size_t)(b * C_ + g * CPG_) * HW_;
  const int tid = threadIdx.x;

  float s = 0.f, ss = 0.f;
  for (int i = tid; i < CPG_ * HW_; i += 256) {
    float v = xb[i];
    s += v; ss += v * v;
  }
  red[0][tid] = s; red[1][tid] = ss;
  __syncthreads();
  for (int o = 128; o > 0; o >>= 1) {
    if (tid < o) { red[0][tid] += red[0][tid + o]; red[1][tid] += red[1][tid + o]; }
    __syncthreads();
  }
  const float inv_n = 1.f / (float)(CPG_ * HW_);
  const float mean = red[0][0] * inv_n;
  const float var  = red[1][0] * inv_n - mean * mean;
  const float rstd = rsqrtf(var + 1e-5f);

  for (int n0 = 0; n0 < HW_; n0 += 256) {
    for (int i = tid; i < CPG_ * 256; i += 256) {
      const int c = i >> 8, n = i & 255;
      const int cg = g * CPG_ + c;
      tile[c][n] = (f16)((xb[(size_t)c * HW_ + n0 + n] - mean) * rstd * gw[cg] + gb[cg]);
    }
    __syncthreads();
    {
      const int n = tid;
      v8h lo, hi;
#pragma unroll
      for (int c = 0; c < 8; ++c)  lo[c] = tile[c][n];
#pragma unroll
      for (int c = 0; c < 8; ++c)  hi[c] = tile[8 + c][n];
      f16* dst = ht + ((size_t)b * HW_ + n0 + n) * C_ + g * CPG_;
      *(v8h*)(dst)     = lo;
      *(v8h*)(dst + 8) = hi;
    }
    __syncthreads();
  }
}

// ---------------------------------------------------------------------------
// Kernel 2: QKV GEMM. Block = 128(o) x 64(n); 8 waves share the async-staged,
// ping-pong double-buffered B tile (64n x 32k N-major in LDS). One barrier
// per K-step; async prefetch of step i+1 overlaps WMMAs of step i.
// Outputs q,k,v stored [b, h, n, d] with 16B stores (q pre-scaled by hd^-0.5).
// ---------------------------------------------------------------------------
__global__ void __launch_bounds__(256) k_qkv_gemm(const f16* __restrict__ wq,
                                                  const float* __restrict__ bias,
                                                  const f16* __restrict__ ht,
                                                  f16* __restrict__ q,
                                                  f16* __restrict__ kk,
                                                  f16* __restrict__ v) {
  __shared__ __align__(16) _Float16 ldsB[2][64 * 32];   // ping-pong, 2 x 4KB
  const int tid = threadIdx.x;
  const int wslot = tid >> 5, lane = tid & 31;
  const int coln = lane & 15, half = lane >> 4;

  const int nb = blockIdx.x & 63;            // 64 n-blocks of 64
  const int ob = (blockIdx.x >> 6) % 12;     // 12 o-blocks of 128
  const int b  = blockIdx.x / (64 * 12);
  const int n0 = nb * 64;
  const int o0 = ob * 128 + wslot * 16;

  const f16* hb = ht + (size_t)b * HW_ * C_;
  const int sn = tid >> 2;                   // staging: n row 0..63
  const int sk = (tid & 3) << 3;             // staging: k seg {0,8,16,24}
  const f16* sg = hb + (size_t)(n0 + sn) * C_ + sk;   // + k0
  f16* const sl0 = &ldsB[0][sn * 32 + sk];
  f16* const sl1 = &ldsB[1][sn * 32 + sk];

  v8f acc[4];
#pragma unroll
  for (int j = 0; j < 4; ++j)
#pragma unroll
    for (int e = 0; e < 8; ++e) acc[j][e] = 0.f;

  async_ld_b128(sg, sl0);                    // prologue: stage K-step 0
  for (int it = 0; it < 16; ++it) {
    const int k0 = it * 32;
    wait_async0();
    __syncthreads();                         // buf[it&1] ready for everyone
    if (it + 1 < 16)                         // prefetch next step (overlapped)
      async_ld_b128(sg + k0 + 32, ((it + 1) & 1) ? sl1 : sl0);
    const f16* bb = ldsB[it & 1];
    const v16h a = ld_a16x32(wq + (size_t)o0 * C_ + k0, C_);
#pragma unroll
    for (int j = 0; j < 4; ++j)
      acc[j] = WMMA_F16(a, ld_b32x16_nmajor(bb + (16 * j) * 32, 32), acc[j]);
  }

  // epilogue: D tile row o = o0 + 8*half + r, col n = n0 + 16j + coln
  const int seg = o0 / C_;                   // 0=q 1=k 2=v
  const int oc  = o0 % C_;
  const int hh  = oc / HD_;
  const int dbase = (oc % HD_) + (half << 3);
  f16* dst = (seg == 0) ? q : (seg == 1) ? kk : v;
  const float sc = (seg == 0) ? 0.125f : 1.f;   // hd^-0.5 folded into q
  float b8[8];
#pragma unroll
  for (int r = 0; r < 8; ++r) b8[r] = bias[o0 + (half << 3) + r];
#pragma unroll
  for (int j = 0; j < 4; ++j) {
    const int n = n0 + 16 * j + coln;
    v8h o8;
#pragma unroll
    for (int r = 0; r < 8; ++r) o8[r] = (f16)((acc[j][r] + b8[r]) * sc);
    *(v8h*)(dst + ((size_t)(b * NH_ + hh) * HW_ + n) * HD_ + dbase) = o8;
  }
}

// ---------------------------------------------------------------------------
// Kernel 3: flash attention. Block = (b, h, 128 queries); 8 waves x 16 queries
// share ping-pong async-staged K/V 32-key chunks. One barrier per chunk; the
// next chunk's async copies overlap S / softmax / PV of the current chunk.
// ---------------------------------------------------------------------------
__global__ void __launch_bounds__(256) k_attention(const f16* __restrict__ q,
                                                   const f16* __restrict__ kk,
                                                   const f16* __restrict__ v,
                                                   f16* __restrict__ ao) {
  __shared__ __align__(16) _Float16 smem[12288];   // 24KB
  f16* const ldsK0 = smem;                         // [32][64] 4KB
  f16* const ldsK1 = smem + 2048;
  f16* const ldsV0 = smem + 4096;                  // [32][64] 4KB
  f16* const ldsV1 = smem + 6144;
  f16* const ldsP  = smem + 8192;                  // 8 x [16][32] 8KB

  const int tid = threadIdx.x;
  const int wslot = tid >> 5, lane = tid & 31;
  const int coln = lane & 15, half = lane >> 4;

  const int bh   = blockIdx.x >> 5;   // 16 (b,h) pairs
  const int qblk = blockIdx.x & 31;   // 32 query blocks of 128
  const int b = bh >> 3, hh = bh & 7;
  const int n0 = qblk * 128 + wslot * 16;

  const f16* qb = q  + (size_t)bh * HW_ * HD_;
  const f16* kb = kk + (size_t)bh * HW_ * HD_;
  const f16* vb = v  + (size_t)bh * HW_ * HD_;

  const v16h aq0 = ld_a16x32(qb + (size_t)n0 * HD_,      HD_);  // d 0..31
  const v16h aq1 = ld_a16x32(qb + (size_t)n0 * HD_ + 32, HD_);  // d 32..63

  v8f acc[4];
  float mi[8], si[8];
#pragma unroll
  for (int t = 0; t < 4; ++t)
#pragma unroll
    for (int e = 0; e < 8; ++e) acc[t][e] = 0.f;
#pragma unroll
  for (int r = 0; r < 8; ++r) { mi[r] = -3.0e38f; si[r] = 0.f; }
  v8f zz;
#pragma unroll
  for (int e = 0; e < 8; ++e) zz[e] = 0.f;

  const int skey = tid >> 3;               // staging: key 0..31
  const int sseg = (tid & 7) << 3;         // staging: d seg {0..56}
  const size_t goff = (size_t)skey * HD_ + sseg;   // + m0*HD_
  const int    loff = skey * HD_ + sseg;
  f16* pslot = ldsP + wslot * 512;

  async_ld_b128(kb + goff, ldsK0 + loff);  // prologue: chunk 0
  async_ld_b128(vb + goff, ldsV0 + loff);
  for (int it = 0; it < 128; ++it) {
    const int m0 = it * 32;
    wait_async0();
    __syncthreads();                       // current chunk resident
    const f16* cK = (it & 1) ? ldsK1 : ldsK0;
    const f16* cV = (it & 1) ? ldsV1 : ldsV0;
    if (it + 1 < 128) {                    // prefetch next chunk (overlapped)
      f16* nK = (it & 1) ? ldsK0 : ldsK1;
      f16* nV = (it & 1) ? ldsV0 : ldsV1;
      async_ld_b128(kb + (size_t)(m0 + 32) * HD_ + goff, nK + loff);
      async_ld_b128(vb + (size_t)(m0 + 32) * HD_ + goff, nV + loff);
    }

    // ---- S = Q K^T : two 16x16 tiles, K=64 chained; K frags N-major in LDS
    v8f s0 = WMMA_F16(aq0, ld_b32x16_nmajor(cK, HD_), zz);
    s0 = WMMA_F16(aq1, ld_b32x16_nmajor(cK + 32, HD_), s0);
    v8f s1 = WMMA_F16(aq0, ld_b32x16_nmajor(cK + 16 * HD_, HD_), zz);
    s1 = WMMA_F16(aq1, ld_b32x16_nmajor(cK + 16 * HD_ + 32, HD_), s1);

    // ---- online softmax over the 32-key chunk ----
#pragma unroll
    for (int r = 0; r < 8; ++r) {
      float t0 = s0[r], t1 = s1[r];
      float cm = fmaxf(t0, t1);
      cm = fmaxf(cm, __shfl_xor(cm, 1, 16));
      cm = fmaxf(cm, __shfl_xor(cm, 2, 16));
      cm = fmaxf(cm, __shfl_xor(cm, 4, 16));
      cm = fmaxf(cm, __shfl_xor(cm, 8, 16));
      const float nm = fmaxf(mi[r], cm);
      const float scl = __expf(mi[r] - nm);
      mi[r] = nm;
      const float p0 = __expf(t0 - nm);
      const float p1 = __expf(t1 - nm);
      float rs = p0 + p1;
      rs += __shfl_xor(rs, 1, 16);
      rs += __shfl_xor(rs, 2, 16);
      rs += __shfl_xor(rs, 4, 16);
      rs += __shfl_xor(rs, 8, 16);
      si[r] = si[r] * scl + rs;
      acc[0][r] *= scl; acc[1][r] *= scl; acc[2][r] *= scl; acc[3][r] *= scl;
      const int row = (half << 3) + r;
      pslot[row * 32 + coln]      = (f16)p0;
      pslot[row * 32 + 16 + coln] = (f16)p1;
    }

    // ---- PV: P (16x32) from LDS as A frag; V frags K-major from LDS ----
    const v16h aP = ld_a16x32(pslot, 32);
#pragma unroll
    for (int t = 0; t < 4; ++t)
      acc[t] = WMMA_F16(aP, ld_b32x16_kmajor(cV + 16 * t, HD_), acc[t]);
  }

  // ---- epilogue: normalize, re-lay N-major via LDS, store [b, n, c] ----
  __syncthreads();
  f16* ot = smem + wslot * 1024;      // per-wave [16 q][64 d]
#pragma unroll
  for (int t = 0; t < 4; ++t)
#pragma unroll
    for (int r = 0; r < 8; ++r)
      ot[((half << 3) + r) * HD_ + t * 16 + coln] = (f16)(acc[t][r] / si[r]);
  if (lane < 16) {
    const int n = n0 + lane;
    f16* dst = ao + ((size_t)b * HW_ + n) * C_ + hh * HD_;
    const f16* src = ot + lane * HD_;
#pragma unroll
    for (int sgp = 0; sgp < 8; ++sgp)
      *(v8h*)(dst + sgp * 8) = *(const v8h*)(src + sgp * 8);
  }
}

// ---------------------------------------------------------------------------
// Kernel 4: proj GEMM + bias + residual; same double-buffered staging.
// ---------------------------------------------------------------------------
__global__ void __launch_bounds__(256) k_proj_gemm(const f16* __restrict__ wp,
                                                   const float* __restrict__ bias,
                                                   const f16* __restrict__ aot,
                                                   const float* __restrict__ x,
                                                   float* __restrict__ out) {
  __shared__ __align__(16) _Float16 ldsB[2][64 * 32];    // 2 x 4KB
  __shared__ __align__(16) float fout[8][16][64];        // 32KB
  const int tid = threadIdx.x;
  const int wslot = tid >> 5, lane = tid & 31;
  const int coln = lane & 15, half = lane >> 4;

  const int nb = blockIdx.x & 63;
  const int ob = (blockIdx.x >> 6) & 3;      // 4 o-blocks of 128
  const int b  = blockIdx.x / 256;
  const int n0 = nb * 64;
  const int o0 = ob * 128 + wslot * 16;

  const f16* ab = aot + (size_t)b * HW_ * C_;
  const int sn = tid >> 2;
  const int sk = (tid & 3) << 3;
  const f16* sg = ab + (size_t)(n0 + sn) * C_ + sk;
  f16* const sl0 = &ldsB[0][sn * 32 + sk];
  f16* const sl1 = &ldsB[1][sn * 32 + sk];

  v8f acc[4];
#pragma unroll
  for (int j = 0; j < 4; ++j)
#pragma unroll
    for (int e = 0; e < 8; ++e) acc[j][e] = 0.f;

  async_ld_b128(sg, sl0);
  for (int it = 0; it < 16; ++it) {
    const int k0 = it * 32;
    wait_async0();
    __syncthreads();
    if (it + 1 < 16)
      async_ld_b128(sg + k0 + 32, ((it + 1) & 1) ? sl1 : sl0);
    const f16* bb = ldsB[it & 1];
    const v16h a = ld_a16x32(wp + (size_t)o0 * C_ + k0, C_);
#pragma unroll
    for (int j = 0; j < 4; ++j)
      acc[j] = WMMA_F16(a, ld_b32x16_nmajor(bb + (16 * j) * 32, 32), acc[j]);
  }

  // stash D tile in LDS (row = o-local, col = n-local), then 256B-row stores
#pragma unroll
  for (int j = 0; j < 4; ++j)
#pragma unroll
    for (int r = 0; r < 8; ++r)
      fout[wslot][(half << 3) + r][16 * j + coln] = acc[j][r];
  if (lane < 16) {
    const int o = o0 + lane;
    const float bo = bias[o];
    const size_t idx = (size_t)(b * C_ + o) * HW_ + n0;
    const float* src = &fout[wslot][lane][0];
#pragma unroll
    for (int n = 0; n < 64; ++n)
      out[idx + n] = x[idx + n] + src[n] + bo;
  }
}

// ---------------------------------------------------------------------------
extern "C" void kernel_launch(void* const* d_in, const int* in_sizes, int n_in,
                              void* d_out, int out_size, void* d_ws, size_t ws_size,
                              hipStream_t stream) {
  (void)in_sizes; (void)n_in; (void)out_size; (void)ws_size;
  const float* x      = (const float*)d_in[0];
  const float* gn_w   = (const float*)d_in[1];
  const float* gn_b   = (const float*)d_in[2];
  const float* qkv_w  = (const float*)d_in[3];
  const float* qkv_b  = (const float*)d_in[4];
  const float* proj_w = (const float*)d_in[5];
  const float* proj_b = (const float*)d_in[6];
  float* out = (float*)d_out;

  char* ws = (char*)d_ws;
  size_t off = 0;
  auto carve = [&](size_t bytes) -> char* {
    char* p = ws + off;
    off += (bytes + 255) & ~(size_t)255;
    return p;
  };
  const size_t act = (size_t)B_ * C_ * HW_;      // 4M elements
  f16* wq = (f16*)carve((size_t)3 * C_ * C_ * sizeof(f16));
  f16* wp = (f16*)carve((size_t)C_ * C_ * sizeof(f16));
  f16* ht = (f16*)carve(act * sizeof(f16));      // [b, n, c]
  f16* q  = (f16*)carve(act * sizeof(f16));      // [b, h, n, d]
  f16* kk = (f16*)carve(act * sizeof(f16));      // [b, h, n, d]
  f16* v  = (f16*)carve(act * sizeof(f16));      // [b, h, n, d]
  f16* ao = (f16*)carve(act * sizeof(f16));      // [b, n, c]

  k_convert_weights<<<256, 256, 0, stream>>>(qkv_w, proj_w, wq, wp);
  k_groupnorm<<<B_ * G_, 256, 0, stream>>>(x, gn_w, gn_b, ht);
  k_qkv_gemm<<<12 * 64 * B_, 256, 0, stream>>>(wq, qkv_b, ht, q, kk, v);
  k_attention<<<16 * 32, 256, 0, stream>>>(q, kk, v, ao);
  k_proj_gemm<<<4 * 64 * B_, 256, 0, stream>>>(wp, proj_b, ao, x, out);
}
